// SpCPE_687194768142
// MI455X (gfx1250) — matrix operand order repaired and checked by reading, not compiled, not run on hardware
//
#include <hip/hip_runtime.h>
#include <hip/hip_bf16.h>

// SubMConv3d(96->96, 27 offsets) + Linear(96->96) + LayerNorm + residual.
// Compute-bound (~52 GFLOP) -> both GEMMs via v_wmma_f32_16x16x32_bf16.
// feat (38 MB) + weights (<1 MB) are L2-resident (192 MB L2). Conv weights are
// double-buffered into LDS with gfx1250 async-to-LDS copies; B fragments are
// software-pipelined (2-deep) so WMMAs don't stall on ds/vmem latency; each
// wave computes a 32x96 tile so every B fragment feeds two WMMAs.

#define C_CH   96
#define K_OFF  27
#define LN_EPS 1e-5f

typedef __attribute__((ext_vector_type(16))) __bf16 v16bf;
typedef __attribute__((ext_vector_type(8)))  float  v8f;
typedef __attribute__((ext_vector_type(4)))  int    v4i;

union FragU {
  uint4 u[2];
  v16bf v;
};

#define ASG __attribute__((address_space(1)))
#define ASL __attribute__((address_space(3)))

#if defined(__gfx1250__) && __has_builtin(__builtin_amdgcn_global_load_async_to_lds_b128)
#define HAVE_ASYNC_LDS 1
#else
#define HAVE_ASYNC_LDS 0
#endif

__device__ __forceinline__ void wait_async() {
#if HAVE_ASYNC_LDS
#if __has_builtin(__builtin_amdgcn_s_wait_asynccnt)
  __builtin_amdgcn_s_wait_asynccnt(0);
#else
  asm volatile("s_wait_asynccnt 0x0" ::: "memory");
#endif
#endif
}

// Copy 18 KB (1152 uint4) of weight fragments global -> LDS, 128 threads.
__device__ __forceinline__ void stage_weights(const uint4* __restrict__ gsrc,
                                              uint4* ldst, int tid) {
#if HAVE_ASYNC_LDS
#pragma unroll
  for (int i = 0; i < 9; ++i) {
    int idx = tid + i * 128;
    __builtin_amdgcn_global_load_async_to_lds_b128(
        (ASG v4i*)(gsrc + idx), (ASL v4i*)(ldst + idx), 0, 0);
  }
#else
  uint4 t[9];
#pragma unroll
  for (int i = 0; i < 9; ++i) t[i] = gsrc[tid + i * 128];
#pragma unroll
  for (int i = 0; i < 9; ++i) ldst[tid + i * 128] = t[i];
#endif
}

// f32x16 (four float4 runs) -> bf16 A-fragment via native cvt (v_cvt_pk_bf16_f32)
__device__ __forceinline__ v16bf pack16(float4 a, float4 b, float4 c, float4 d) {
  v16bf v;
  v[0]  = (__bf16)a.x; v[1]  = (__bf16)a.y; v[2]  = (__bf16)a.z; v[3]  = (__bf16)a.w;
  v[4]  = (__bf16)b.x; v[5]  = (__bf16)b.y; v[6]  = (__bf16)b.z; v[7]  = (__bf16)b.w;
  v[8]  = (__bf16)c.x; v[9]  = (__bf16)c.y; v[10] = (__bf16)c.z; v[11] = (__bf16)c.w;
  v[12] = (__bf16)d.x; v[13] = (__bf16)d.y; v[14] = (__bf16)d.z; v[15] = (__bf16)d.w;
  return v;
}

// Gather one voxel row's 96 channels (two 32B runs per K-chunk) as 3 A-frags.
__device__ __forceinline__ void gather_a(const float* __restrict__ feat,
                                         int nb, int N, int half, v16bf afr[3]) {
  bool nv = (unsigned)nb < (unsigned)N;
  const float* fr = feat + (size_t)(nv ? nb : 0) * C_CH;
#pragma unroll
  for (int kc = 0; kc < 3; ++kc) {
    float4 r0a = {0.f, 0.f, 0.f, 0.f}, r0b = r0a, r1a = r0a, r1b = r0a;
    if (nv) {
      const float4* p0 = (const float4*)(fr + kc * 32 + half * 8);
      const float4* p1 = (const float4*)(fr + kc * 32 + 16 + half * 8);
      r0a = p0[0]; r0b = p0[1];
      r1a = p1[0]; r1b = p1[1];
    }
    afr[kc] = pack16(r0a, r0b, r1a, r1b);
  }
}

// ---------------------------------------------------------------------------
// Prep: convert W_conv [27,96,96] f32 and W_lin [96,96] f32 (used transposed)
// into bf16 B-fragments, fragment-ordered: each lane's 16 elements contiguous.
// B-fragment layout (32x16 bf16, wave32, ISA 7.12.2):
//   lane: n = lane&15, half = lane>>4 ; element j: B[16*(j>>3) + half*8 + (j&7)][n]
// Conv fragment id f = (k*3 + kc)*6 + nt ; linear fragments at f >= 486.
// Storage: frag[f][lane][j]  (512 bf16 per fragment = 1 KB)
// ---------------------------------------------------------------------------
__global__ __launch_bounds__(128) void prep_weights_kernel(
    const float* __restrict__ Wc, const float* __restrict__ Wl,
    __bf16* __restrict__ frag) {
  const int NCONV = K_OFF * 3 * 6;          // 486
  const int NFRAG = NCONV + 3 * 6;          // 504
  int t = blockIdx.x * blockDim.x + threadIdx.x;
  if (t >= NFRAG * 32) return;
  int f = t >> 5, lane = t & 31;
  int n15 = lane & 15, half = lane >> 4;
  __bf16* dst = frag + (size_t)f * 512 + lane * 16;
  if (f < NCONV) {
    int k = f / 18, rem = f % 18, kc = rem / 6, nt = rem % 6;
    const float* W = Wc + (size_t)k * C_CH * C_CH;   // B[c][d] = W_conv[k][c][d]
#pragma unroll
    for (int j = 0; j < 16; ++j) {
      int c = kc * 32 + 16 * (j >> 3) + half * 8 + (j & 7);
      dst[j] = (__bf16)W[(size_t)c * C_CH + nt * 16 + n15];
    }
  } else {
    int f2 = f - NCONV;
    int kc = f2 / 6, nt = f2 % 6;                    // B[c][d] = W_lin[d][c]
#pragma unroll
    for (int j = 0; j < 16; ++j) {
      int c = kc * 32 + 16 * (j >> 3) + half * 8 + (j & 7);
      dst[j] = (__bf16)Wl[(size_t)(nt * 16 + n15) * C_CH + c];
    }
  }
}

// ---------------------------------------------------------------------------
// Main: 128 threads = 4 waves; each wave owns a 32-voxel x 96-channel tile
// (two 16-row A sets, 12 f32 accumulators). Per offset: 18 B fragments from
// LDS, each feeding 2 WMMAs, with a 2-deep register pipeline on B.
// A-fragment layout (16x32 bf16, ISA 7.12.2):
//   lane: m = lane&15, half = lane>>4 ; element j: A[m][16*(j>>3)+half*8+(j&7)]
// C/D layout: lane n = lane&15 ; VGPR r -> row m = r + 8*(lane>>4)
// ---------------------------------------------------------------------------
__global__ __launch_bounds__(128) void spconv_ln_kernel(
    const float* __restrict__ feat,
    const int*   __restrict__ nbr,
    const float* __restrict__ b_conv,
    const float* __restrict__ b_lin,
    const float* __restrict__ gamma,
    const float* __restrict__ beta,
    const __bf16* __restrict__ wfrag,
    float* __restrict__ out,
    int N) {
  __shared__ __align__(16) uint4  lds_w[2][18 * 64];        // 2 x 18 KB weights
  __shared__ __align__(16) __bf16 lds_conv[4][32 * C_CH];   // 4 x 6 KB transpose

  const int tid  = threadIdx.x;
  const int wave = tid >> 5;
  const int lane = tid & 31;
  const int n15  = lane & 15;
  const int half = lane >> 4;
  const int base = (blockIdx.x * 4 + wave) * 32;
  const int grow0 = base + n15;        // rows 0..15 of this wave's tile
  const int grow1 = grow0 + 16;        // rows 16..31

  const uint4* wsrc = (const uint4*)wfrag;   // 64 uint4 per fragment

  const v8f vzero = {0.f, 0.f, 0.f, 0.f, 0.f, 0.f, 0.f, 0.f};
  v8f acc0[6], acc1[6];
#pragma unroll
  for (int nt = 0; nt < 6; ++nt) { acc0[nt] = vzero; acc1[nt] = vzero; }

  // prologue: stage k=0 weights
  stage_weights(wsrc, &lds_w[0][0], tid);
  __builtin_prefetch(wfrag + (size_t)(K_OFF * 18) * 512, 0, 0);  // linear weights
  wait_async();
  __syncthreads();

  // ---- submanifold conv: sum_k gather(feat, nbr[:,k]) @ W_conv[k] ----
  for (int k = 0; k < K_OFF; ++k) {
    const uint4* wbuf = &lds_w[k & 1][0];
    if (k + 1 < K_OFF)   // async copy of next offset's weights overlaps compute
      stage_weights(wsrc + (size_t)(k + 1) * 1152, &lds_w[(k + 1) & 1][0], tid);

    int nb0 = (grow0 < N) ? nbr[grow0 * K_OFF + k] : N;   // pad row index == N
    int nb1 = (grow1 < N) ? nbr[grow1 * K_OFF + k] : N;
    v16bf afr0[3], afr1[3];
    gather_a(feat, nb0, N, half, afr0);
    gather_a(feat, nb1, N, half, afr1);

    // 18 B fragments, 2-deep pipeline, each feeds both row groups
    FragU fb[2];
    {
      const uint4* p = wbuf + lane * 2;   // ord 0 -> (nt=0,kc=0) offset 0
      fb[0].u[0] = p[0]; fb[0].u[1] = p[1];
    }
#pragma unroll
    for (int ord = 0; ord < 18; ++ord) {
      const int nt = ord / 3, kc = ord % 3;
      if (ord + 1 < 18) {
        const int nn = (ord + 1) / 3, nk = (ord + 1) % 3;
        const uint4* p = wbuf + (nk * 6 + nn) * 64 + lane * 2;
        fb[(ord + 1) & 1].u[0] = p[0];
        fb[(ord + 1) & 1].u[1] = p[1];
      }
      acc0[nt] = __builtin_amdgcn_wmma_f32_16x16x32_bf16(
          false, afr0[kc], false, fb[ord & 1].v, (short)0, acc0[nt], false, false);
      acc1[nt] = __builtin_amdgcn_wmma_f32_16x16x32_bf16(
          false, afr1[kc], false, fb[ord & 1].v, (short)0, acc1[nt], false, false);
    }

    wait_async();     // next-k weights landed in LDS
    __syncthreads();  // all waves done reading wbuf before it is overwritten
  }

  // ---- conv bias, stage conv result through LDS (C/D layout -> A layout) ----
#pragma unroll
  for (int nt = 0; nt < 6; ++nt) {
    float bc = b_conv[nt * 16 + n15];
#pragma unroll
    for (int r = 0; r < 8; ++r) {
      int m = r + 8 * half;
      lds_conv[wave][m * C_CH + nt * 16 + n15]        = (__bf16)(acc0[nt][r] + bc);
      lds_conv[wave][(16 + m) * C_CH + nt * 16 + n15] = (__bf16)(acc1[nt][r] + bc);
    }
  }
  __syncthreads();

  v16bf ca0[3], ca1[3];
#pragma unroll
  for (int kc = 0; kc < 3; ++kc) {  // 16B-aligned ds_load_b128 pairs
    FragU fa;
    fa.u[0] = *(const uint4*)(&lds_conv[wave][n15 * C_CH + kc * 32 + half * 8]);
    fa.u[1] = *(const uint4*)(&lds_conv[wave][n15 * C_CH + kc * 32 + 16 + half * 8]);
    ca0[kc] = fa.v;
    fa.u[0] = *(const uint4*)(&lds_conv[wave][(16 + n15) * C_CH + kc * 32 + half * 8]);
    fa.u[1] = *(const uint4*)(&lds_conv[wave][(16 + n15) * C_CH + kc * 32 + 16 + half * 8]);
    ca1[kc] = fa.v;
  }

  // ---- linear: conv @ W_lin^T (weight fragments from L2, 2-deep pipeline) ----
  v8f lacc0[6], lacc1[6];
#pragma unroll
  for (int nt = 0; nt < 6; ++nt) { lacc0[nt] = vzero; lacc1[nt] = vzero; }
  const uint4* wl = (const uint4*)(wfrag + (size_t)(K_OFF * 18) * 512);
  FragU fb[2];
  {
    const uint4* p = wl + lane * 2;
    fb[0].u[0] = p[0]; fb[0].u[1] = p[1];
  }
#pragma unroll
  for (int ord = 0; ord < 18; ++ord) {
    const int nt = ord / 3, kc = ord % 3;
    if (ord + 1 < 18) {
      const int nn = (ord + 1) / 3, nk = (ord + 1) % 3;
      const uint4* p = wl + (nk * 6 + nn) * 64 + lane * 2;
      fb[(ord + 1) & 1].u[0] = p[0];
      fb[(ord + 1) & 1].u[1] = p[1];
    }
    lacc0[nt] = __builtin_amdgcn_wmma_f32_16x16x32_bf16(
        false, ca0[kc], false, fb[ord & 1].v, (short)0, lacc0[nt], false, false);
    lacc1[nt] = __builtin_amdgcn_wmma_f32_16x16x32_bf16(
        false, ca1[kc], false, fb[ord & 1].v, (short)0, lacc1[nt], false, false);
  }

  // ---- bias + LayerNorm (rows live in one 16-lane half) + residual ----
  float g[6], bt[6], bl[6];
#pragma unroll
  for (int nt = 0; nt < 6; ++nt) {
    int d = nt * 16 + n15;
    g[nt] = gamma[d]; bt[nt] = beta[d]; bl[nt] = b_lin[d];
  }

#pragma unroll
  for (int grp = 0; grp < 2; ++grp) {
#pragma unroll
    for (int r = 0; r < 8; ++r) {
      float s = 0.f, sq = 0.f;
#pragma unroll
      for (int nt = 0; nt < 6; ++nt) {
        float x = (grp ? lacc1[nt][r] : lacc0[nt][r]) + bl[nt];
        s += x; sq += x * x;
      }
#pragma unroll
      for (int m = 1; m <= 8; m <<= 1) {  // reduce across the 16-lane half
        s  += __shfl_xor(s,  m, 32);
        sq += __shfl_xor(sq, m, 32);
      }
      float mu  = s * (1.f / 96.f);
      float var = sq * (1.f / 96.f) - mu * mu;
      float inv = rsqrtf(var + LN_EPS);
      int row = base + grp * 16 + r + 8 * half;
      if (row < N) {
        const float* fp = feat + (size_t)row * C_CH;
        float*       op = out  + (size_t)row * C_CH;
#pragma unroll
        for (int nt = 0; nt < 6; ++nt) {
          int d = nt * 16 + n15;
          float x = (grp ? lacc1[nt][r] : lacc0[nt][r]) + bl[nt];
          op[d] = fp[d] + (x - mu) * inv * g[nt] + bt[nt];
        }
      }
    }
  }
}

// ---------------------------------------------------------------------------
extern "C" void kernel_launch(void* const* d_in, const int* in_sizes, int n_in,
                              void* d_out, int out_size, void* d_ws, size_t ws_size,
                              hipStream_t stream) {
  const float* feat   = (const float*)d_in[0];
  const int*   nbr    = (const int*)  d_in[1];
  const float* W_conv = (const float*)d_in[2];
  const float* b_conv = (const float*)d_in[3];
  const float* W_lin  = (const float*)d_in[4];
  const float* b_lin  = (const float*)d_in[5];
  const float* gamma  = (const float*)d_in[6];
  const float* beta   = (const float*)d_in[7];

  const int N = in_sizes[0] / C_CH;

  // workspace: 504 fragments * 1 KB = 516096 bytes of pre-swizzled bf16 weights
  __bf16* wfrag = (__bf16*)d_ws;
  const int nfrag_threads = (K_OFF * 18 + 18) * 32;  // 16128
  prep_weights_kernel<<<(nfrag_threads + 127) / 128, 128, 0, stream>>>(
      W_conv, W_lin, wfrag);

  const int blocks = (N + 127) / 128;  // 4 waves/block, 32 voxels per wave
  spconv_ln_kernel<<<blocks, 128, 0, stream>>>(
      feat, nbr, b_conv, b_lin, gamma, beta, wfrag, (float*)d_out, N);
}
